// MyModel_47313359733329
// MI455X (gfx1250) — matrix operand order repaired and checked by reading, not compiled, not run on hardware
//
#include <hip/hip_runtime.h>
#include <stdint.h>

#define HH 384
#define WW 384
#define HWSZ (HH * WW)
#define PW (WW + 2)              // padded grid width
#define PH (HH + 2)
#define PSZ (PW * PH)
#define KC 8
#define IMS 4                    // padded input stride: ch0-2 = img, ch3 = 0
#define CATS 20                  // concat stride: ch0-2 = a, ch3-18 = att, ch19 = 0

typedef _Float16 v16h __attribute__((ext_vector_type(16)));
typedef _Float16 h8   __attribute__((ext_vector_type(8)));
typedef _Float16 h4   __attribute__((ext_vector_type(4)));
typedef float    v8f  __attribute__((ext_vector_type(8)));

// ---------------------------------------------------------------------------
// helpers
// ---------------------------------------------------------------------------
__device__ __forceinline__ uint32_t hash32(uint32_t x) {
    x ^= x >> 16; x *= 0x7feb352dU;
    x ^= x >> 15; x *= 0x846ca68bU;
    x ^= x >> 16;
    return x;
}
__device__ __forceinline__ int clampi(int v, int lo, int hi) {
    return v < lo ? lo : (v > hi ? hi : v);
}

// A-matrix 16x32 f16 fragment layout (ISA 7.12.2):
//  lane l, element i (i = 2*vgpr + half):  M = l&15
//  VGPR v: group = v>>2, K = group*16 + ((l>=16)?8:0) + (v&3)*2 + half
__device__ __forceinline__ void decodeA(int l, int i, int& M, int& K) {
    int v = i >> 1, h = i & 1;
    M = l & 15;
    int grp = v >> 2;
    K = grp * 16 + ((l >> 4) ? 8 : 0) + ((v & 3) << 1) + h;
}

// B-element constant offset on a padded pixel-major grid.
// K -> (c, tap); tap -> (dy,dx); invalid K -> zero pad channel.
__device__ __forceinline__ constexpr int bOff(int K, int Kused, int stride, int padCh) {
    int c = (K < Kused) ? (K / 9) : padCh;
    int t = K % 9;
    int dy = t / 3 - 1, dx = t % 3 - 1;
    return (dy * PW + dx) * stride + c;
}

// ---------------------------------------------------------------------------
// Kernel 0: build padded f16 pixel-major copies of a and b (ch3 = 0), and zero
// the border of the concat buffer (interior written later by combine).
// ---------------------------------------------------------------------------
__global__ void pad_inputs(const float* __restrict__ a, const float* __restrict__ b,
                           _Float16* __restrict__ aP, _Float16* __restrict__ bP,
                           _Float16* __restrict__ catP) {
    int idx = blockIdx.x * blockDim.x + threadIdx.x;
    if (idx >= PSZ) return;
    int py = idx / PW, px = idx % PW;
    bool interior = (py >= 1) & (py <= HH) & (px >= 1) & (px <= WW);
    h4 av = {}, bv = {};
    if (interior) {
        int p = (py - 1) * WW + (px - 1);
        av[0] = (_Float16)a[0 * HWSZ + p]; bv[0] = (_Float16)b[0 * HWSZ + p];
        av[1] = (_Float16)a[1 * HWSZ + p]; bv[1] = (_Float16)b[1 * HWSZ + p];
        av[2] = (_Float16)a[2 * HWSZ + p]; bv[2] = (_Float16)b[2 * HWSZ + p];
    } else {
        _Float16* cp = catP + (size_t)idx * CATS;   // zero concat border
#pragma unroll
        for (int j = 0; j < CATS; j += 4) *(h4*)(cp + j) = h4{};
    }
    *(h4*)(aP + (size_t)idx * IMS) = av;
    *(h4*)(bP + (size_t)idx * IMS) = bv;
}

// ---------------------------------------------------------------------------
// Kernel 1: pack conv weights into WMMA A-fragment order (f32 -> f16).
//   wq/wk/wv: (16,3,3,3) OIHW -> K = c*9+tap (27 used, pad to 32)
//   wf:       (3,19,3,3)      -> K = c*9+tap (171 used, pad to 192 = 6 chunks)
// packed[l*16+i] so each lane loads its v16h fragment with one 32B read.
// ---------------------------------------------------------------------------
__global__ void pack_weights(const float* __restrict__ wq, const float* __restrict__ wk,
                             const float* __restrict__ wv, const float* __restrict__ wf,
                             _Float16* pq, _Float16* pk, _Float16* pv, _Float16* pf) {
    int t = threadIdx.x;
    for (int idx = t; idx < 512; idx += 256) {
        int l = idx >> 4, i = idx & 15;
        int M, K; decodeA(l, i, M, K);
        float q = 0.f, k = 0.f, v = 0.f;
        if (K < 27) { int w = M * 27 + K; q = wq[w]; k = wk[w]; v = wv[w]; }
        pq[idx] = (_Float16)q; pk[idx] = (_Float16)k; pv[idx] = (_Float16)v;
    }
    for (int idx = t; idx < 6 * 512; idx += 256) {
        int chunk = idx >> 9, rem = idx & 511;
        int l = rem >> 4, i = rem & 15;
        int M, K; decodeA(l, i, M, K);
        int Kg = chunk * 32 + K;
        float val = 0.f;
        if (M < 3 && Kg < 171) val = wf[M * 171 + Kg];
        pf[idx] = (_Float16)val;
    }
}

// ---------------------------------------------------------------------------
// Kernel 2: 3x3 conv 3->16 + bias + ReLU via implicit-GEMM WMMA.
// One wave = one 16ch x 16px tile = ONE v_wmma_f32_16x16x32_f16.
// B gathers: base + compile-time constant offset on the padded 4-ch image.
// ---------------------------------------------------------------------------
template <bool OUTF16>
__global__ void conv16_wmma(const _Float16* __restrict__ imgP,
                            const _Float16* __restrict__ packedA,
                            const float* __restrict__ bias, void* __restrict__ dstv,
                            int ntiles) {
    int wave = threadIdx.x >> 5;
    int lane = threadIdx.x & 31;
    int tile = blockIdx.x * (blockDim.x >> 5) + wave;
    if (tile >= ntiles) return;                 // wave-uniform: EXEC stays full

    v16h afrag = *(const v16h*)(packedA + lane * 16);

    int N = lane & 15;
    int p = tile * 16 + N;
    int y = p / WW, x = p % WW;
    bool hi = lane >= 16;
    int base = ((y + 1) * PW + (x + 1)) * IMS;  // center pixel in padded grid

    v16h bfrag;
#pragma unroll
    for (int i = 0; i < 16; i++) {
        const int offA = bOff(i < 8 ? i : i + 8,      27, IMS, 3);   // lanes 0-15  (K = i<8 ? i : i+8)
        const int offB = bOff(i < 8 ? i + 8 : i + 16, 27, IMS, 3);   // lanes 16-31 (K = i<8 ? i+8 : i+16)
        int off = hi ? offB : offA;                  // one cndmask
        bfrag[i] = imgP[base + off];                 // always in-bounds
    }

    v8f acc = {};
    acc = __builtin_amdgcn_wmma_f32_16x16x32_f16(false, afrag, false, bfrag,
                                                 (short)0, acc, false, false);

    int mbase = hi ? 8 : 0;                     // C: VGPR v <-> M = mbase + v
    if (OUTF16) {
        h8 o;
#pragma unroll
        for (int v = 0; v < 8; v++) o[v] = (_Float16)fmaxf(acc[v] + bias[mbase + v], 0.f);
        *(h8*)((_Float16*)dstv + p * 16 + mbase) = o;      // one 16B store
    } else {
        float* d = (float*)dstv + p * 16 + mbase;
        float4 o0, o1;
        o0.x = fmaxf(acc[0] + bias[mbase + 0], 0.f);
        o0.y = fmaxf(acc[1] + bias[mbase + 1], 0.f);
        o0.z = fmaxf(acc[2] + bias[mbase + 2], 0.f);
        o0.w = fmaxf(acc[3] + bias[mbase + 3], 0.f);
        o1.x = fmaxf(acc[4] + bias[mbase + 4], 0.f);
        o1.y = fmaxf(acc[5] + bias[mbase + 5], 0.f);
        o1.z = fmaxf(acc[6] + bias[mbase + 6], 0.f);
        o1.w = fmaxf(acc[7] + bias[mbase + 7], 0.f);
        *(float4*)d = o0; *(float4*)(d + 4) = o1;          // two 16B stores
    }
}

// ---------------------------------------------------------------------------
// PatchMatch cost: 16-ch squared distance, q in registers, k gathered (32B, L2)
// ---------------------------------------------------------------------------
__device__ __forceinline__ float costEval(v16h q, const _Float16* __restrict__ kf,
                                          int cy, int cx) {
    v16h k = *(const v16h*)(kf + (cy * WW + cx) * 16);
    float s = 0.f;
#pragma unroll
    for (int c = 0; c < 16; c++) {
        float d = (float)q[c] - (float)k[c];
        s = fmaf(d, d, s);
    }
    return s;
}

// Kernel 3: init 8 random candidates per pixel. State pixel-major: 32B/thread.
__global__ void pm_init(const _Float16* __restrict__ qf, const _Float16* __restrict__ kf,
                        float* __restrict__ cost, uint32_t* __restrict__ pos) {
    int p = blockIdx.x * blockDim.x + threadIdx.x;
    if (p >= HWSZ) return;
    v16h q = *(const v16h*)(qf + p * 16);
#pragma unroll
    for (int i = 0; i < KC; i++) {
        uint32_t hy = hash32((uint32_t)p * 8u + i + 0x01234567u);
        uint32_t hx = hash32((uint32_t)p * 8u + i + 0x89abcdefu);
        int cy = (int)(hy % HH), cx = (int)(hx % WW);
        cost[p * KC + i] = costEval(q, kf, cy, cx);
        pos[p * KC + i]  = ((uint32_t)cy << 16) | (uint32_t)cx;
    }
}

// Kernel 4: one propagation step (ping-pong; reads neighbors' pre-update state).
//  mode 0: horizontal (dir=+1: left neighbor, x+1 | dir=-1: right neighbor, x-1)
//  mode 1: vertical   (dir=+1: up neighbor,  y+1 | dir=-1: down neighbor,  y-1)
__global__ void pm_prop(const _Float16* __restrict__ qf, const _Float16* __restrict__ kf,
                        const float* __restrict__ costIn, const uint32_t* __restrict__ posIn,
                        float* __restrict__ costOut, uint32_t* __restrict__ posOut,
                        int mode, int dir) {
    int p = blockIdx.x * blockDim.x + threadIdx.x;
    if (p >= HWSZ) return;
    int y = p / WW, x = p % WW;
    v16h q = *(const v16h*)(qf + p * 16);

    int np = p;                                  // neighbor (self at edges, as in ref)
    if (mode == 0) { int nx = x - dir; if ((unsigned)nx < (unsigned)WW) np = y * WW + nx; }
    else           { int ny = y - dir; if ((unsigned)ny < (unsigned)HH) np = ny * WW + x; }

    float bc[KC]; uint32_t bp[KC], npos[KC];
#pragma unroll
    for (int i = 0; i < KC; i++) {               // contiguous -> b128 loads
        bc[i] = costIn[p * KC + i];
        bp[i] = posIn[p * KC + i];
        npos[i] = posIn[np * KC + i];
    }
#pragma unroll
    for (int i = 0; i < KC; i++) {
        int cy = (int)(npos[i] >> 16), cx = (int)(npos[i] & 0xffff);
        if (mode == 0) cx += dir; else cy += dir;
        cy = clampi(cy, 0, HH - 1); cx = clampi(cx, 0, WW - 1);
        float cc = costEval(q, kf, cy, cx);
        if (cc < bc[i]) { bc[i] = cc; bp[i] = ((uint32_t)cy << 16) | (uint32_t)cx; }
    }
#pragma unroll
    for (int i = 0; i < KC; i++) {
        costOut[p * KC + i] = bc[i];
        posOut[p * KC + i]  = bp[i];
    }
}

// Kernel 5: all FOUR random-search rounds of one iteration, fused, in-place.
// Each thread reads/writes only its own 64B of state -> race-free in-place.
__global__ void pm_random4(const _Float16* __restrict__ qf, const _Float16* __restrict__ kf,
                           float* __restrict__ cost, uint32_t* __restrict__ pos, uint32_t salt) {
    int p = blockIdx.x * blockDim.x + threadIdx.x;
    if (p >= HWSZ) return;
    v16h q = *(const v16h*)(qf + p * 16);

    float bc[KC]; uint32_t bp[KC];
#pragma unroll
    for (int i = 0; i < KC; i++) { bc[i] = cost[p * KC + i]; bp[i] = pos[p * KC + i]; }

    const int radii[4] = {WW / 2, WW / 8, WW / 32, 4};   // 192, 48, 12, 4
#pragma unroll
    for (int r = 0; r < 4; r++) {
        const int rad = radii[r];
        const int span = 2 * rad + 1;                    // compile-time modulus
        uint32_t s1 = salt + (uint32_t)r * 0x01000193u;
#pragma unroll
        for (int i = 0; i < KC; i++) {
            uint32_t hy = hash32(((uint32_t)p * 8u + i) ^ s1);
            uint32_t hx = hash32(((uint32_t)p * 8u + i) ^ (s1 * 0x9e3779b9u + 0x5bd1e995u));
            int cy = (int)(bp[i] >> 16)    + (int)(hy % (uint32_t)span) - rad;
            int cx = (int)(bp[i] & 0xffff) + (int)(hx % (uint32_t)span) - rad;
            cy = clampi(cy, 0, HH - 1); cx = clampi(cx, 0, WW - 1);
            float cc = costEval(q, kf, cy, cx);
            if (cc < bc[i]) { bc[i] = cc; bp[i] = ((uint32_t)cy << 16) | (uint32_t)cx; }
        }
    }
#pragma unroll
    for (int i = 0; i < KC; i++) { cost[p * KC + i] = bc[i]; pos[p * KC + i] = bp[i]; }
}

// Kernel 6: softmax(-cost) over 8 candidates + weighted gather of v; writes the
// interior of the padded concat buffer catP[pix][20] (ch0-2 = a, ch3-18 = att).
__global__ void combine(const float* __restrict__ cost, const uint32_t* __restrict__ pos,
                        const float* __restrict__ vf, const _Float16* __restrict__ aP,
                        _Float16* __restrict__ catP) {
    int p = blockIdx.x * blockDim.x + threadIdx.x;
    if (p >= HWSZ) return;
    float c[KC]; uint32_t pp[KC];
    float cmin = 3.402823e38f;
#pragma unroll
    for (int i = 0; i < KC; i++) {
        c[i] = cost[p * KC + i];
        pp[i] = pos[p * KC + i];
        cmin = fminf(cmin, c[i]);
    }
    float w[KC], wsum = 0.f;
#pragma unroll
    for (int i = 0; i < KC; i++) { w[i] = __expf(cmin - c[i]); wsum += w[i]; }
    float inv = 1.f / wsum;

    float accv[16];
#pragma unroll
    for (int ch = 0; ch < 16; ch++) accv[ch] = 0.f;
    for (int i = 0; i < KC; i++) {
        int p2 = (int)(pp[i] >> 16) * WW + (int)(pp[i] & 0xffff);
        const float4* vp = (const float4*)(vf + p2 * 16);
        float wi = w[i] * inv;
#pragma unroll
        for (int j = 0; j < 4; j++) {
            float4 vv = vp[j];
            accv[j * 4 + 0] = fmaf(wi, vv.x, accv[j * 4 + 0]);
            accv[j * 4 + 1] = fmaf(wi, vv.y, accv[j * 4 + 1]);
            accv[j * 4 + 2] = fmaf(wi, vv.z, accv[j * 4 + 2]);
            accv[j * 4 + 3] = fmaf(wi, vv.w, accv[j * 4 + 3]);
        }
    }
    int y = p / WW, x = p % WW;
    size_t pidx = (size_t)((y + 1) * PW + (x + 1));
    _Float16 outv[CATS];
    const _Float16* ap = aP + pidx * IMS;
    outv[0] = ap[0]; outv[1] = ap[1]; outv[2] = ap[2];
#pragma unroll
    for (int ch = 0; ch < 16; ch++) outv[3 + ch] = (_Float16)accv[ch];
    outv[19] = (_Float16)0.f;
    _Float16* cp = catP + pidx * CATS;
#pragma unroll
    for (int j = 0; j < CATS; j += 4) *(h4*)(cp + j) = *(h4*)(outv + j);  // 8B stores
}

// ---------------------------------------------------------------------------
// Kernel 7: final 3x3 conv 19->3 (+bias, sigmoid) via 6 chained WMMAs (K=192).
// B gathers: base + compile-time constant offset on the padded concat buffer.
// ---------------------------------------------------------------------------
__global__ void conv_final_wmma(const _Float16* __restrict__ catP,
                                const _Float16* __restrict__ packedA,
                                const float* __restrict__ bias,
                                float* __restrict__ out, int ntiles) {
    int wave = threadIdx.x >> 5;
    int lane = threadIdx.x & 31;
    int tile = blockIdx.x * (blockDim.x >> 5) + wave;
    if (tile >= ntiles) return;                  // wave-uniform

    int N = lane & 15;
    int p = tile * 16 + N;
    int y = p / WW, x = p % WW;
    bool hi = lane >= 16;
    int base = ((y + 1) * PW + (x + 1)) * CATS;

    v8f acc = {};
#pragma unroll
    for (int chunk = 0; chunk < 6; chunk++) {
        v16h afrag = *(const v16h*)(packedA + chunk * 512 + lane * 16);
        v16h bfrag;
#pragma unroll
        for (int i = 0; i < 16; i++) {
            const int offA = bOff(chunk * 32 + (i < 8 ? i : i + 8),      171, CATS, 19);
            const int offB = bOff(chunk * 32 + (i < 8 ? i + 8 : i + 16), 171, CATS, 19);
            int off = hi ? offB : offA;
            bfrag[i] = catP[base + off];
        }
        acc = __builtin_amdgcn_wmma_f32_16x16x32_f16(false, afrag, false, bfrag,
                                                     (short)0, acc, false, false);
    }

    int mbase = hi ? 8 : 0;
#pragma unroll
    for (int v = 0; v < 8; v++) {
        int m = mbase + v;
        if (m < 3) {                             // only out-channels 0..2 are real
            float o = acc[v] + bias[m];
            o = 1.f / (1.f + __expf(-o));
            out[m * HWSZ + p] = o;               // NCHW planar output
        }
    }
}

// ---------------------------------------------------------------------------
// Host launcher
// ---------------------------------------------------------------------------
extern "C" void kernel_launch(void* const* d_in, const int* in_sizes, int n_in,
                              void* d_out, int out_size, void* d_ws, size_t ws_size,
                              hipStream_t stream) {
    (void)in_sizes; (void)n_in; (void)out_size; (void)ws_size;
    const float* a  = (const float*)d_in[0];
    const float* b  = (const float*)d_in[1];
    const float* wq = (const float*)d_in[2]; const float* bq = (const float*)d_in[3];
    const float* wk = (const float*)d_in[4]; const float* bk = (const float*)d_in[5];
    const float* wv = (const float*)d_in[6]; const float* bv = (const float*)d_in[7];
    const float* wf = (const float*)d_in[8]; const float* bf = (const float*)d_in[9];

    char* ws = (char*)d_ws;
    size_t off = 0;
    auto alloc = [&](size_t bytes) -> void* {
        void* pt = ws + off;
        off = (off + bytes + 255) & ~(size_t)255;
        return pt;
    };
    _Float16* aP   = (_Float16*)alloc((size_t)PSZ * IMS * 2);   // padded a, f16
    _Float16* bP   = (_Float16*)alloc((size_t)PSZ * IMS * 2);   // padded b, f16
    _Float16* qf   = (_Float16*)alloc((size_t)HWSZ * 16 * 2);   // q, f16 pixel-major
    _Float16* kfh  = (_Float16*)alloc((size_t)HWSZ * 16 * 2);   // k, f16 pixel-major
    float*    vf   = (float*)   alloc((size_t)HWSZ * 16 * 4);   // v, f32 pixel-major
    _Float16* catP = (_Float16*)alloc((size_t)PSZ * CATS * 2);  // padded concat, f16
    float*    cost0= (float*)   alloc((size_t)HWSZ * KC * 4);
    float*    cost1= (float*)   alloc((size_t)HWSZ * KC * 4);
    uint32_t* pos0 = (uint32_t*)alloc((size_t)HWSZ * KC * 4);
    uint32_t* pos1 = (uint32_t*)alloc((size_t)HWSZ * KC * 4);
    _Float16* pq   = (_Float16*)alloc(512 * 2);
    _Float16* pk   = (_Float16*)alloc(512 * 2);
    _Float16* pv   = (_Float16*)alloc(512 * 2);
    _Float16* pf   = (_Float16*)alloc(6 * 512 * 2);

    pad_inputs<<<(PSZ + 255) / 256, 256, 0, stream>>>(a, b, aP, bP, catP);
    pack_weights<<<1, 256, 0, stream>>>(wq, wk, wv, wf, pq, pk, pv, pf);

    const int ntiles = HWSZ / 16;                // 9216 tiles of 16 pixels
    const int cblocks = ntiles / 8;              // 8 waves / 256-thread block
    conv16_wmma<true ><<<cblocks, 256, 0, stream>>>(aP, pq, bq, (void*)qf,  ntiles);
    conv16_wmma<true ><<<cblocks, 256, 0, stream>>>(bP, pk, bk, (void*)kfh, ntiles);
    conv16_wmma<false><<<cblocks, 256, 0, stream>>>(bP, pv, bv, (void*)vf,  ntiles);

    const int pblocks = HWSZ / 256;              // 576
    pm_init<<<pblocks, 256, 0, stream>>>(qf, kfh, cost0, pos0);

    // Per iteration: prop-h (S0->S1), prop-v (S1->S0), fused random x4 (S0 in place)
    for (int it = 0; it < 5; it++) {
        int dir = (it % 2 == 0) ? 1 : -1;
        pm_prop<<<pblocks, 256, 0, stream>>>(qf, kfh, cost0, pos0, cost1, pos1, 0, dir);
        pm_prop<<<pblocks, 256, 0, stream>>>(qf, kfh, cost1, pos1, cost0, pos0, 1, dir);
        pm_random4<<<pblocks, 256, 0, stream>>>(qf, kfh, cost0, pos0,
                                                0x30000u + (uint32_t)it * 0x101u);
    }

    combine<<<pblocks, 256, 0, stream>>>(cost0, pos0, vf, aP, catP);

    conv_final_wmma<<<cblocks, 256, 0, stream>>>(catP, pf, bf, (float*)d_out, ntiles);
}
